// PatchClassifier_25623774888147
// MI455X (gfx1250) — matrix operand order repaired and checked by reading, not compiled, not run on hardware
//
#include <hip/hip_runtime.h>
#include <stdint.h>

// PatchClassifier: logits = node @ patch^T ; softmax ; recon = probs @ patch ;
// probs_ = onehot(argmax) - probs + probs (straight-through).
// N=100000, P=512, D=256. bf16 WMMA (v_wmma_f32_16x16x32_bf16) for both GEMMs.

#define NN 100000
#define PP 512
#define DD 256

typedef __bf16 bf16_t;
typedef bf16_t v16bf __attribute__((ext_vector_type(16)));
typedef float  v8f   __attribute__((ext_vector_type(8)));

union Frag {
  v16bf v;
  unsigned short u[16];
};

__device__ __forceinline__ unsigned short f32_to_bf16(float f) {
  unsigned int u = __float_as_uint(f);
  u += 0x7fffu + ((u >> 16) & 1u);   // round-to-nearest-even
  return (unsigned short)(u >> 16);
}

// ---------------------------------------------------------------------------
// Pack patch_emb [P=512][D=256] f32 -> bf16 WMMA B-fragments in d_ws.
// Region B1 (GEMM1, B = patch^T, K=D, N=P): 32 n-tiles x 8 k-steps, frag=1KB.
// Region B2 (GEMM2, B = patch,   K=P, N=D): 16 n-tiles x 16 k-steps.
// B fragment layout (16-bit B 32x16): lane l -> column n = l%16,
// half h -> K = (l/16)*8 + h + (h/8)*8  (ISA order; 16 halfs per lane = 32B).
// ---------------------------------------------------------------------------
__global__ void pack_patch(const float* __restrict__ patch,
                           unsigned short* __restrict__ ws) {
  const int tid  = blockIdx.x * blockDim.x + threadIdx.x;  // 0..131071
  const int h    = tid & 15;
  const int l    = (tid >> 4) & 31;
  const int frag = tid >> 9;                                // 0..255
  const int g = l >> 4, ln = l & 15;
  {
    const int j = frag >> 3, ks = frag & 7;                 // j: P-tile, ks: D-step
    const int p = j * 16 + ln;
    const int d = ks * 32 + g * 8 + h + ((h >> 3) << 3);
    ws[(size_t)frag * 512 + l * 16 + h] = f32_to_bf16(patch[p * DD + d]);
  }
  {
    const int j2 = frag >> 4, ks2 = frag & 15;              // j2: D-tile, ks2: P-step
    const int d = j2 * 16 + ln;
    const int p = ks2 * 32 + g * 8 + h + ((h >> 3) << 3);
    ws[131072 + (size_t)frag * 512 + l * 16 + h] = f32_to_bf16(patch[p * DD + d]);
  }
}

// ---------------------------------------------------------------------------
// Main kernel: 128 threads = 4 waves, each wave owns 16 rows.
// ---------------------------------------------------------------------------
__global__ __launch_bounds__(128) void patch_cls_main(
    const float* __restrict__ node,
    const unsigned short* __restrict__ wsB1,
    const unsigned short* __restrict__ wsB2,
    float* __restrict__ probsOut,   // [N, 512]
    float* __restrict__ nodeOut)    // [N, 256]
{
  constexpr int LSTR = 520;  // logits row stride (floats), padded vs 512
  constexpr int PSTR = 520;  // probs  row stride (bf16 halfs), 16B-aligned rows
  extern __shared__ char smem[];

  const int wave = threadIdx.x >> 5;
  const int lane = threadIdx.x & 31;
  const int g  = lane >> 4;
  const int ln = lane & 15;

  float*          logitsLds = (float*)smem + wave * 16 * LSTR;
  unsigned short* probsLds  =
      (unsigned short*)(smem + 4 * 16 * LSTR * 4) + wave * 16 * PSTR;

  const int rowBase = blockIdx.x * 64 + wave * 16;
  int arow = rowBase + ln;
  if (arow > NN - 1) arow = NN - 1;                 // clamp loads; mask stores
  const float* arowp = node + (size_t)arow * DD;

  // ---- A1 fragments: 16x32 bf16 per k-step (ISA 16-bit A layout),
  //      halfs 0..7 -> K = k0+8g+h ; halfs 8..15 -> K = k0+16+8g+(h-8)
  Frag a1[8];
  #pragma unroll
  for (int ks = 0; ks < 8; ++ks) {
    const int k0 = ks * 32;
    const float4 f0 = *(const float4*)(arowp + k0 + 8 * g);
    const float4 f1 = *(const float4*)(arowp + k0 + 8 * g + 4);
    const float4 f2 = *(const float4*)(arowp + k0 + 16 + 8 * g);
    const float4 f3 = *(const float4*)(arowp + k0 + 16 + 8 * g + 4);
    const float fl[16] = {f0.x, f0.y, f0.z, f0.w, f1.x, f1.y, f1.z, f1.w,
                          f2.x, f2.y, f2.z, f2.w, f3.x, f3.y, f3.z, f3.w};
    #pragma unroll
    for (int h = 0; h < 16; ++h) a1[ks].u[h] = f32_to_bf16(fl[h]);
  }

  // ---- GEMM1: logits[16 x 512] = A1[16x256] * B1[256x512], 4 N-tiles/chunk
  for (int chunk = 0; chunk < 8; ++chunk) {
    v8f c[4] = {};
    #pragma unroll
    for (int ks = 0; ks < 8; ++ks) {
      #pragma unroll
      for (int t = 0; t < 4; ++t) {
        const int j = chunk * 4 + t;
        const v16bf b =
            *(const v16bf*)(wsB1 + (size_t)(j * 8 + ks) * 512 + lane * 16);
        c[t] = __builtin_amdgcn_wmma_f32_16x16x32_bf16(
            false, a1[ks].v, false, b, (short)0, c[t], false, false);
      }
    }
    // C layout: VGPR r -> row (r + 8g), col = ln within tile
    #pragma unroll
    for (int t = 0; t < 4; ++t)
      #pragma unroll
      for (int r = 0; r < 8; ++r)
        logitsLds[(r + 8 * g) * LSTR + (chunk * 4 + t) * 16 + ln] = c[t][r];
  }

  // ---- Softmax + argmax per row (fp32, wave32 shuffle reductions) ----
  for (int r = 0; r < 16; ++r) {
    const float* rowp = logitsLds + r * LSTR;
    float vals[16];
    float m = -3.4e38f;
    #pragma unroll
    for (int i = 0; i < 16; ++i) {
      vals[i] = rowp[lane + 32 * i];
      m = fmaxf(m, vals[i]);
    }
    #pragma unroll
    for (int off = 16; off > 0; off >>= 1) m = fmaxf(m, __shfl_xor(m, off, 32));

    float s = 0.f;
    float bv = -3.4e38f;
    int   bi = 0x7fffffff;
    #pragma unroll
    for (int i = 0; i < 16; ++i) {
      const int p = lane + 32 * i;
      if (vals[i] > bv) { bv = vals[i]; bi = p; }   // first-max tie break (i asc)
      const float e = __expf(vals[i] - m);
      vals[i] = e;
      s += e;
    }
    #pragma unroll
    for (int off = 16; off > 0; off >>= 1) s += __shfl_xor(s, off, 32);
    #pragma unroll
    for (int off = 16; off > 0; off >>= 1) {
      const float ov = __shfl_xor(bv, off, 32);
      const int   oi = __shfl_xor(bi, off, 32);
      if (ov > bv || (ov == bv && oi < bi)) { bv = ov; bi = oi; }
    }
    const float invs = 1.f / s;
    const int   n    = rowBase + r;
    const bool  act  = n < NN;
    float* op = probsOut + (size_t)n * PP;
    #pragma unroll
    for (int i = 0; i < 16; ++i) {
      const int   p    = lane + 32 * i;
      const float pr   = vals[i] * invs;
      const float hard = (p == bi) ? 1.f : 0.f;
      if (act) op[p] = hard - pr + pr;              // straight-through expr
      probsLds[r * PSTR + p] = f32_to_bf16(pr);     // A of GEMM2
    }
  }

  // ---- GEMM2: recon[16 x 256] = probs[16x512] * B2[512x256] ----
  for (int chunk = 0; chunk < 4; ++chunk) {
    v8f c2[4] = {};
    #pragma unroll
    for (int ks = 0; ks < 16; ++ks) {
      const int k0 = ks * 32;
      Frag a2;
      *(float4*)&a2.u[0] =
          *(const float4*)(probsLds + ln * PSTR + k0 + 8 * g);
      *(float4*)&a2.u[8] =
          *(const float4*)(probsLds + ln * PSTR + k0 + 16 + 8 * g);
      #pragma unroll
      for (int t = 0; t < 4; ++t) {
        const int j2 = chunk * 4 + t;
        const v16bf b2 =
            *(const v16bf*)(wsB2 + (size_t)(j2 * 16 + ks) * 512 + lane * 16);
        c2[t] = __builtin_amdgcn_wmma_f32_16x16x32_bf16(
            false, a2.v, false, b2, (short)0, c2[t], false, false);
      }
    }
    #pragma unroll
    for (int t = 0; t < 4; ++t) {
      const int col = (chunk * 4 + t) * 16 + ln;
      #pragma unroll
      for (int r = 0; r < 8; ++r) {
        const int n = rowBase + r + 8 * g;
        if (n < NN) nodeOut[(size_t)n * DD + col] = c2[t][r];
      }
    }
  }
}

extern "C" void kernel_launch(void* const* d_in, const int* in_sizes, int n_in,
                              void* d_out, int out_size, void* d_ws,
                              size_t ws_size, hipStream_t stream) {
  (void)in_sizes; (void)n_in; (void)out_size; (void)ws_size;
  const float* node  = (const float*)d_in[0];   // [N, D] f32
  const float* patch = (const float*)d_in[1];   // [P, D] f32
  unsigned short* ws = (unsigned short*)d_ws;   // 512 KB bf16 fragments
  float* probsOut = (float*)d_out;              // [N, P]
  float* nodeOut  = probsOut + (size_t)NN * PP; // [N, D]

  // Pack patch_emb into WMMA B fragments (L2-resident, 512 KB).
  pack_patch<<<512, 256, 0, stream>>>(patch, ws);

  // Main fused kernel: 64 rows per block, 4 waves.
  const int smemBytes = 4 * 16 * 520 * 4 + 4 * 16 * 520 * 2;  // ~195 KB
  patch_cls_main<<<(NN + 63) / 64, 128, smemBytes, stream>>>(
      node, ws, ws + 131072, probsOut, nodeOut);
}
